// TLSTMMotionForecaster_79946521247954
// MI455X (gfx1250) — 8-bit matrix operand order repaired and checked by reading, compiled, not run on hardware
//
#include <hip/hip_runtime.h>
#include <math.h>

typedef int   v16i __attribute__((ext_vector_type(16)));
typedef float v8f  __attribute__((ext_vector_type(8)));

#define ROWS 16
#define HT   128
#define LSEQ 64
#define DIN  16
#define FOUT 60
#define NFRAG_G1 96
#define NFRAG_WD 16
#define FRAG_BYTES 2048
#define PACK_BYTES ((NFRAG_G1 + NFRAG_WD) * FRAG_BYTES)

__device__ __forceinline__ unsigned char f32_to_e4m3(float x) {
  unsigned int u = __float_as_uint(x);
  unsigned int s = (u >> 24) & 0x80u;
  float ax = fabsf(x);
  unsigned int au = u & 0x7FFFFFFFu;
  int E = (int)((au >> 23) & 0xFFu) - 127;
  unsigned int mr = (au & 0x7FFFFFu) + 0x80000u;
  unsigned int m3 = mr >> 20;
  E += (int)(m3 >> 3);
  m3 &= 7u;
  unsigned int normv = s | ((unsigned int)(E + 7) << 3) | m3;
  unsigned int subq  = s | (unsigned int)(int)(ax * 512.0f + 0.5f);
  unsigned int r = (ax < 0.015625f) ? subq : normv;
  r = (ax >= 448.0f) ? (s | 0x7Eu) : r;
  r = (x != x) ? 0x7Fu : r;
  return (unsigned char)r;
}

__device__ __forceinline__ float fast_rcp(float x) { return __builtin_amdgcn_rcpf(x); }
__device__ __forceinline__ float sigm(float x) { return fast_rcp(1.0f + __expf(-x)); }
__device__ __forceinline__ float tanh_fast(float x) {
  return 1.0f - 2.0f * fast_rcp(1.0f + __expf(2.0f * x));
}

__device__ __forceinline__ void a8_store(unsigned char* base, int m, int k, unsigned char val) {
  int group = k >> 5;
  int k32   = k & 31;
  int hl    = (k32 >> 3) & 1;
  int vv    = (((k32 >> 4) & 1) << 1) | ((k32 >> 2) & 1);
  int v     = group * 4 + vv;
  base[(hl * 16 + m) * 64 + v * 4 + (k & 3)] = val;
}

__global__ __launch_bounds__(256) void pack_weights(const float* __restrict__ Wx,
                                                    const float* __restrict__ Uh,
                                                    const float* __restrict__ Wd,
                                                    unsigned char* __restrict__ wpk) {
  int gid = blockIdx.x * 256 + threadIdx.x;
  if (gid >= PACK_BYTES) return;
  int fragid  = gid >> 11;
  int within  = gid & 2047;
  int lane    = within >> 6;
  int byteoff = within & 63;
  int v  = byteoff >> 2;
  int bb = byteoff & 3;
  int hl = lane >> 4;
  int n  = lane & 15;
  int k  = (v >> 2) * 32 + ((v >> 1) & 1) * 16 + hl * 8 + (v & 1) * 4 + bb;
  float src;
  if (fragid < NFRAG_G1) {
    int kf  = fragid & 1;
    int tmp = fragid >> 1;
    int nt  = tmp & 7;
    int t2  = tmp >> 3;
    int g   = t2 % 3;
    int l   = t2 / 3;
    int K   = kf * 128 + k;
    int col = nt * 16 + n;
    if (K < 128) src = Wx[((l * 4 + (g + 1)) * HT + K) * HT + col];
    else         src = Uh[((l * 4 + (g + 1)) * HT + (K - 128)) * HT + col];
  } else {
    int f2  = fragid - NFRAG_G1;
    int l   = f2 >> 3;
    int nt  = f2 & 7;
    int col = nt * 16 + n;
    src = Wd[(l * HT + k) * HT + col];
  }
  wpk[gid] = f32_to_e4m3(src);
}

__global__ __launch_bounds__(256) void tlstm_main(
    const float* __restrict__ hist, const float* __restrict__ msk,
    const float* __restrict__ Wp,   const float* __restrict__ bp,
    const float* __restrict__ bx,   const float* __restrict__ bd,
    const float* __restrict__ Wt,   const float* __restrict__ bt,
    const float* __restrict__ ln_g, const float* __restrict__ ln_b,
    const float* __restrict__ W1,   const float* __restrict__ b1,
    const float* __restrict__ W2,   const float* __restrict__ b2,
    const unsigned char* __restrict__ wpk, float* __restrict__ out) {

  __shared__ __attribute__((aligned(64))) int A8[2][32][16];
  __shared__ __attribute__((aligned(64))) int A2[32][16];
  __shared__ float WpL[DIN][HT];
  __shared__ float bpL[HT];
  __shared__ float histT[ROWS][DIN];
  __shared__ float decayL[ROWS];
  __shared__ float mtL[ROWS];
  __shared__ int   lastL[ROWS];
  __shared__ float henc[ROWS][HT];
  __shared__ float snL[ROWS][144];
  __shared__ float p1L[ROWS][HT];

  const int tid  = threadIdx.x;
  const int lane = tid & 31;
  const int w    = tid >> 5;
  const int hl   = (lane >> 4) & 1;
  const int nn   = lane & 15;
  const int col  = w * 16 + nn;
  const int rowbase = blockIdx.x * ROWS;

  for (int i = tid; i < DIN * HT; i += 256) WpL[i >> 7][i & 127] = Wp[i];
  for (int i = tid; i < HT; i += 256) bpL[i] = bp[i];

  float bI[2], bO[2], bC[2], bdS[2], wtS[2], btS[2];
#pragma unroll
  for (int l = 0; l < 2; ++l) {
    bI[l]  = bx[(l * 4 + 1) * HT + col];
    bO[l]  = bx[(l * 4 + 2) * HT + col];
    bC[l]  = bx[(l * 4 + 3) * HT + col];
    bdS[l] = bd[l * HT + col];
    wtS[l] = Wt[l * HT + col];
    btS[l] = bt[l * HT + col];
  }

  v16i wG[2][3][2];
  v16i wD[2];
#pragma unroll
  for (int l = 0; l < 2; ++l) {
#pragma unroll
    for (int g = 0; g < 3; ++g)
#pragma unroll
      for (int kf = 0; kf < 2; ++kf) {
        int fid = ((l * 3 + g) * 8 + w) * 2 + kf;
        wG[l][g][kf] = *(const v16i*)(wpk + fid * FRAG_BYTES + lane * 64);
      }
    wD[l] = *(const v16i*)(wpk + (NFRAG_G1 + l * 8 + w) * FRAG_BYTES + lane * 64);
  }

  if (tid < ROWS) {
    float s = 0.f;
    for (int t = 0; t < LSEQ; ++t) s += msk[(rowbase + tid) * LSEQ + t];
    int cnt = (int)(s + 0.5f);
    cnt = cnt < 1 ? 1 : (cnt > LSEQ ? LSEQ : cnt);
    lastL[tid] = cnt - 1;
  }

  v8f zerov = {};
  v8f h_s[2], c_s[2];
  h_s[0] = zerov; h_s[1] = zerov; c_s[0] = zerov; c_s[1] = zerov;

  __syncthreads();

  for (int t = 0; t < LSEQ; ++t) {
    {
      int m = tid >> 4, k = tid & 15;
      histT[m][k] = hist[((rowbase + m) * LSEQ + t) * DIN + k];
    }
    {
      unsigned char* a1b = (unsigned char*)&A8[1][0][0];
#pragma unroll
      for (int r = 0; r < 8; ++r)
        a8_store(a1b, r + 8 * hl, col, f32_to_e4m3(h_s[0][r]));
    }
    __syncthreads();
    if (tid < ROWS) {
      float dlt = histT[tid][5];
      if (dlt < 0.f) dlt = 0.f;
      decayL[tid] = fast_rcp(__logf(2.718281828f + dlt));
      mtL[tid] = msk[(rowbase + tid) * LSEQ + t];
    }
    {
      int m = tid >> 4, cb = (tid & 15) * 8;
      unsigned char* a0b = (unsigned char*)&A8[0][0][0];
#pragma unroll
      for (int j = 0; j < 8; ++j) {
        int c = cb + j;
        float acc = bpL[c];
#pragma unroll
        for (int k = 0; k < DIN; ++k) acc += histT[m][k] * WpL[k][c];
        a8_store(a0b, m, c, f32_to_e4m3(acc));
      }
    }
    __syncthreads();

#pragma unroll
    for (int l = 0; l < 2; ++l) {
      v16i a0 = *(const v16i*)&A8[0][lane][0];
      v16i a1 = *(const v16i*)&A8[1][lane][0];
      v8f zI = {}, zO = {}, zC = {};
      zI = __builtin_amdgcn_wmma_f32_16x16x128_fp8_fp8(a0, wG[l][0][0], (short)0, zI, false, false);
      zO = __builtin_amdgcn_wmma_f32_16x16x128_fp8_fp8(a0, wG[l][1][0], (short)0, zO, false, false);
      zC = __builtin_amdgcn_wmma_f32_16x16x128_fp8_fp8(a0, wG[l][2][0], (short)0, zC, false, false);
      zI = __builtin_amdgcn_wmma_f32_16x16x128_fp8_fp8(a1, wG[l][0][1], (short)0, zI, false, false);
      zO = __builtin_amdgcn_wmma_f32_16x16x128_fp8_fp8(a1, wG[l][1][1], (short)0, zO, false, false);
      zC = __builtin_amdgcn_wmma_f32_16x16x128_fp8_fp8(a1, wG[l][2][1], (short)0, zC, false, false);

      float o_t[8], h_t[8];
      unsigned char* a2b = (unsigned char*)&A2[0][0];
#pragma unroll
      for (int r = 0; r < 8; ++r) {
        float it = sigm(zI[r] + bI[l]);
        o_t[r]   = sigm(zO[r] + bO[l]);
        float ct = tanh_fast(zC[r] + bC[l]);
        h_t[r] = ct + it;
        a8_store(a2b, r + 8 * hl, col, f32_to_e4m3(h_t[r]));
      }
      __syncthreads();

      v16i a2 = *(const v16i*)&A2[lane][0];
      v8f zS = {};
      zS = __builtin_amdgcn_wmma_f32_16x16x128_fp8_fp8(a2, wD[l], (short)0, zS, false, false);

#pragma unroll
      for (int r = 0; r < 8; ++r) {
        int m = r + 8 * hl;
        float hshort = tanh_fast(zS[r] + bdS[l]);
        float d      = sigm(decayL[m] * wtS[l] + btS[l]);
        float hstar  = (h_t[r] - hshort) + hshort * d;
        float cnew   = tanh_fast(hstar + o_t[r] * c_s[l][r]);
        float hnew   = o_t[r] * tanh_fast(cnew);
        float mt     = mtL[m];
        h_s[l][r] = mt * hnew + (1.f - mt) * h_s[l][r];
        c_s[l][r] = mt * cnew + (1.f - mt) * c_s[l][r];
      }
      if (l == 0) {
        unsigned char* a0b = (unsigned char*)&A8[0][0][0];
        unsigned char* a1b = (unsigned char*)&A8[1][0][0];
#pragma unroll
        for (int r = 0; r < 8; ++r) {
          a8_store(a0b, r + 8 * hl, col, f32_to_e4m3(h_s[0][r]));
          a8_store(a1b, r + 8 * hl, col, f32_to_e4m3(h_s[1][r]));
        }
      } else {
#pragma unroll
        for (int r = 0; r < 8; ++r) {
          int m = r + 8 * hl;
          if (t == lastL[m]) henc[m][col] = h_s[1][r] * mtL[m];
        }
      }
      __syncthreads();
    }
  }

  __syncthreads();
  if (tid < ROWS) {
    int m = tid;
    int lt = lastL[m];
    const float* hrow = hist + ((rowbase + m) * LSEQ + lt) * DIN;
    float s1 = 0.f, s2 = 0.f;
    for (int k = 0; k < DIN; ++k) { float v = hrow[k];    s1 += v; s2 += v * v; }
    for (int k = 0; k < HT;  ++k) { float v = henc[m][k]; s1 += v; s2 += v * v; }
    float mu  = s1 * (1.0f / 144.0f);
    float var = s2 * (1.0f / 144.0f) - mu * mu;
    float inv = rsqrtf(var + 1e-5f);
    for (int k = 0; k < DIN; ++k) snL[m][k]       = (hrow[k]    - mu) * inv * ln_g[k]      + ln_b[k];
    for (int k = 0; k < HT;  ++k) snL[m][DIN + k] = (henc[m][k] - mu) * inv * ln_g[DIN + k] + ln_b[DIN + k];
  }
  __syncthreads();
  {
    int m = tid >> 4, cb = (tid & 15) * 8;
    float acc[8];
#pragma unroll
    for (int j = 0; j < 8; ++j) acc[j] = b1[cb + j];
    for (int k = 0; k < 144; ++k) {
      float sv = snL[m][k];
#pragma unroll
      for (int j = 0; j < 8; ++j) acc[j] += sv * W1[k * HT + cb + j];
    }
#pragma unroll
    for (int j = 0; j < 8; ++j) p1L[m][cb + j] = acc[j] > 0.f ? acc[j] : 0.f;
  }
  __syncthreads();
  for (int idx = tid; idx < ROWS * FOUT; idx += 256) {
    int m = idx / FOUT, o = idx % FOUT;
    float acc = b2[o];
    for (int k = 0; k < HT; ++k) acc += p1L[m][k] * W2[k * FOUT + o];
    if (acc != acc) acc = 0.f;
    else if (__builtin_isinf(acc)) acc = acc > 0.f ? 1e4f : -1e4f;
    out[(rowbase + m) * FOUT + o] = acc;
  }
}

extern "C" void kernel_launch(void* const* d_in, const int* in_sizes, int n_in,
                              void* d_out, int out_size, void* d_ws, size_t ws_size,
                              hipStream_t stream) {
  const float* hist = (const float*)d_in[0];
  const float* msk  = (const float*)d_in[1];
  const float* Wp   = (const float*)d_in[2];
  const float* bp   = (const float*)d_in[3];
  const float* Wx   = (const float*)d_in[4];
  const float* bx   = (const float*)d_in[5];
  const float* Uh   = (const float*)d_in[6];
  const float* Wd   = (const float*)d_in[7];
  const float* bd   = (const float*)d_in[8];
  const float* Wt   = (const float*)d_in[9];
  const float* bt   = (const float*)d_in[10];
  const float* ln_g = (const float*)d_in[11];
  const float* ln_b = (const float*)d_in[12];
  const float* W1   = (const float*)d_in[13];
  const float* b1   = (const float*)d_in[14];
  const float* W2   = (const float*)d_in[15];
  const float* b2   = (const float*)d_in[16];
  unsigned char* wpk = (unsigned char*)d_ws;

  pack_weights<<<(PACK_BYTES + 255) / 256, 256, 0, stream>>>(Wx, Uh, Wd, wpk);
  tlstm_main<<<4096 / ROWS, 256, 0, stream>>>(hist, msk, Wp, bp, bx, bd, Wt, bt,
                                              ln_g, ln_b, W1, b1, W2, b2, wpk,
                                              (float*)d_out);
}
